// dVBF_55087250538936
// MI455X (gfx1250) — compile-verified, matrix operand order checked by reading, not run on hardware
//
#include <hip/hip_runtime.h>
#include <hip/hip_bf16.h>
#include <math.h>

// dims
#define S_ 8
#define B_ 64
#define T_ 1000
#define N_ 128
#define L_ 32
#define H_ 128
#define BT_ (B_ * T_)            // 64000
#define BTL_ (B_ * T_ * L_)      // 2048000
#define SBTL_ (S_ * B_ * T_ * L_)// 16384000

#ifdef __has_builtin
#if __has_builtin(__builtin_amdgcn_global_load_async_to_lds_b128)
#define HAVE_ASYNC_LDS 1
#endif
#endif
#ifndef HAVE_ASYNC_LDS
#define HAVE_ASYNC_LDS 0
#endif

typedef __attribute__((ext_vector_type(8)))  _Float16 v8h;
typedef __attribute__((ext_vector_type(16))) _Float16 v16h;
typedef __attribute__((ext_vector_type(8)))  float    v8f;
typedef __attribute__((ext_vector_type(4)))  int      v4i;

__device__ __forceinline__ v8f wmma32(v16h a, v16h b, v8f c) {
  // D = A(16x32 f16) * B(32x16 f16) + C(16x16 f32)
  return __builtin_amdgcn_wmma_f32_16x16x32_f16(false, a, false, b, (short)0, c,
                                                false, false);
}

// ISA 7.12.2 16-bit A/B fragment K mapping: half j (0..15), lane-half g (0/1):
//   j<8  -> K = j      + 8*g
//   j>=8 -> K = j + 8  + 8*g
// Per lane the fragment is two contiguous 8-f16 runs at +8g and +16+8g.
__device__ __forceinline__ v16h frag_from_lds_row(const _Float16* rowp, int g) {
  v8h lo = *(const v8h*)(rowp + 8 * g);
  v8h hi = *(const v8h*)(rowp + 16 + 8 * g);
  return __builtin_shufflevector(lo, hi, 0, 1, 2, 3, 4, 5, 6, 7,
                                 8, 9, 10, 11, 12, 13, 14, 15);
}

// Gather a 32x16 B fragment (column = 16*n + (lane&15)) from a row-major
// f32 weight matrix W[K x ld], K chunk starting at kbase.
__device__ __forceinline__ v16h bfrag_global(const float* __restrict__ W,
                                             int ld, int kbase, int col, int g) {
  v16h f;
#pragma unroll
  for (int j = 0; j < 16; ++j) {
    int k = kbase + ((j < 8) ? j : j + 8) + 8 * g;
    f[j] = (_Float16)W[k * ld + col];
  }
  return f;
}

__device__ __forceinline__ float softplusf(float x) {
  return (x > 20.f) ? x : log1pf(expf(x));
}

#if HAVE_ASYNC_LDS
typedef v4i __attribute__((address_space(1)))* as1_v4i_p;  // global int4*
typedef v4i __attribute__((address_space(3)))* as3_v4i_p;  // LDS int4*

__device__ __forceinline__ void async_cp16(const float* g, float* l) {
  __builtin_amdgcn_global_load_async_to_lds_b128((as1_v4i_p)g, (as3_v4i_p)l, 0,
                                                 0);
}
#endif

__global__ void k_init_loss(float* loss) {
  if (threadIdx.x == 0 && blockIdx.x == 0) *loss = 0.f;
}

// ---------------- Kernel 1: encoder  y@[Wm|WP] -> m, sqrtP, KL ----------------
// 4000 row tiles of 16, one wave per tile. K=128 -> 4 WMMA chunks, 4 N-tiles.
__global__ __launch_bounds__(128) void k_encoder(
    const float* __restrict__ y, const float* __restrict__ tmask,
    const float* __restrict__ Wm, const float* __restrict__ bm,
    const float* __restrict__ WP, const float* __restrict__ bP,
    float* __restrict__ m_ws, float* __restrict__ sp_ws,
    float* __restrict__ loss) {
  __shared__ _Float16 yst[4][16][128];
  const int lane = threadIdx.x & 31;
  const int wv = threadIdx.x >> 5;
  const int g = lane >> 4;
  const int cl = lane & 15;
  const int tile = blockIdx.x * 4 + wv;  // 0..3999
  const int R0 = tile * 16;              // base row in flattened [B*T]

  // Weight fragments: n=0,1 -> mean head cols 0..31 ; n=2,3 -> var head
  v16h bf[4][4];
#pragma unroll
  for (int kc = 0; kc < 4; ++kc)
#pragma unroll
    for (int n = 0; n < 4; ++n) {
      const float* W = (n < 2) ? Wm : WP;
      bf[kc][n] = bfrag_global(W, L_, 32 * kc, 16 * (n & 1) + cl, g);
    }

  v8f acc[4];
#pragma unroll
  for (int n = 0; n < 4; ++n) {
    float bias = (n < 2) ? bm[16 * (n & 1) + cl] : bP[16 * (n & 1) + cl];
#pragma unroll
    for (int i = 0; i < 8; ++i) acc[n][i] = bias;
  }

  // Stage y tile [16][128] f32 -> f16 LDS
#pragma unroll 8
  for (int i = 0; i < 64; ++i) {
    int idx = lane + 32 * i;
    yst[wv][idx >> 7][idx & 127] = (_Float16)y[(size_t)R0 * N_ + idx];
  }
  asm volatile("s_wait_dscnt 0" ::: "memory");

#pragma unroll
  for (int kc = 0; kc < 4; ++kc) {
    v16h a = frag_from_lds_row(&yst[wv][cl][32 * kc], g);
#pragma unroll
    for (int n = 0; n < 4; ++n) acc[n] = wmma32(a, bf[kc][n], acc[n]);
  }

  float maskv[8];
#pragma unroll
  for (int v = 0; v < 8; ++v) maskv[v] = tmask[R0 + v + 8 * g];

  float klp = 0.f;
#pragma unroll
  for (int n = 0; n < 2; ++n) {
    int col = 16 * n + cl;
#pragma unroll
    for (int v = 0; v < 8; ++v) {
      int row = R0 + v + 8 * g;
      float mk = acc[n][v] * maskv[v];
      float sp = softplusf(acc[n + 2][v]);
      float P = sp * maskv[v] + (1.f - maskv[v]);
      m_ws[(size_t)row * L_ + col] = mk;
      sp_ws[(size_t)row * L_ + col] = sqrtf(P);
      klp += 0.5f * (P + mk * mk - logf(P) - 1.f);
    }
  }
#pragma unroll
  for (int o = 16; o > 0; o >>= 1) klp += __shfl_xor(klp, o, 32);
  if (lane == 0) atomicAdd(loss, klp * (1.0f / B_));
}

// ---------------- Kernel 2: sequential dynamics scan ----------------
// 32 waves total; wave owns 16 chains (fixed s, 16 consecutive b) for all T.
// Noise (m, sqrtP, eps) for step t+1 is fetched with async global->LDS B128
// copies (ASYNCcnt) while step t's WMMA chain executes; double-buffered.
__global__ __launch_bounds__(64) void k_scan(
    const float* __restrict__ eps, const float* __restrict__ W1,
    const float* __restrict__ b1, const float* __restrict__ W2,
    const float* __restrict__ b2, const float* __restrict__ logQ,
    const float* __restrict__ m0, const float* __restrict__ logQ0,
    const float* __restrict__ m_ws, const float* __restrict__ sp_ws,
    float* __restrict__ zout) {
  __shared__ _Float16 zst[2][16][32];   // 2 KB
  __shared__ _Float16 hst[2][16][128];  // 8 KB
#if HAVE_ASYNC_LDS
  __shared__ float nbuf[2][2][3][16][32];  // [wave][buf][m/sp/eps] 24 KB
#endif
  const int lane = threadIdx.x & 31;
  const int wv = threadIdx.x >> 5;
  const int g = lane >> 4;
  const int cl = lane & 15;
  const int tile = blockIdx.x * 2 + wv;  // 0..31
  const int s = tile >> 2;               // sample
  const int b0 = (tile & 3) * 16;        // trial base

  // W1 [32][128]: K=32, 8 N-tiles. W2 [128][32]: 4 K-chunks x 2 N-tiles.
  v16h w1f[8];
#pragma unroll
  for (int n = 0; n < 8; ++n) w1f[n] = bfrag_global(W1, H_, 0, 16 * n + cl, g);
  v16h w2f[4][2];
#pragma unroll
  for (int kc = 0; kc < 4; ++kc)
#pragma unroll
    for (int n = 0; n < 2; ++n)
      w2f[kc][n] = bfrag_global(W2, L_, 32 * kc, 16 * n + cl, g);

  float b1v[8];
#pragma unroll
  for (int n = 0; n < 8; ++n) b1v[n] = b1[16 * n + cl];
  float b2v[2], Qs[2], m0v[2], Q0s[2];
#pragma unroll
  for (int n = 0; n < 2; ++n) {
    int col = 16 * n + cl;
    b2v[n] = b2[col];
    Qs[n] = sqrtf(softplusf(logQ[col]));
    m0v[n] = m0[col];
    Q0s[n] = sqrtf(softplusf(logQ0[col]));
  }

#if HAVE_ASYNC_LDS
  // Issue 12 async B128 copies filling nbuf[wv][pb] with (m, sqrtP, eps)
  // for timestep t: 16 rows x 32 f32 per array, 32 lanes x 16B x 4 groups.
  auto issue_noise = [&](int t, int pb) {
#pragma unroll
    for (int q = 0; q < 4; ++q) {
      int ch = lane + 32 * q;  // 0..127
      int row = ch >> 3;
      int c0 = (ch & 7) * 4;
      int b = b0 + row;
      size_t bt = (size_t)b * T_ + t;
      size_t sbt = ((size_t)(s * B_ + b)) * T_ + t;
      async_cp16(m_ws + bt * L_ + c0, &nbuf[wv][pb][0][row][c0]);
      async_cp16(sp_ws + bt * L_ + c0, &nbuf[wv][pb][1][row][c0]);
      async_cp16(eps + sbt * L_ + c0, &nbuf[wv][pb][2][row][c0]);
    }
  };
#endif

  v8f zacc[2];
  // t = 0: z0 = m_0 + sqrt(softplus(log_Q_0)) * (m + sqrtP*eps)
#pragma unroll
  for (int n = 0; n < 2; ++n) {
    int col = 16 * n + cl;
#pragma unroll
    for (int v = 0; v < 8; ++v) {
      int b = b0 + v + 8 * g;
      size_t bt = (size_t)b * T_;
      size_t sbt = ((size_t)(s * B_ + b)) * T_;
      float w = m_ws[bt * L_ + col] + sp_ws[bt * L_ + col] * eps[sbt * L_ + col];
      float z = m0v[n] + Q0s[n] * w;
      zacc[n][v] = z;
      zout[sbt * L_ + col] = z;
    }
  }

#if HAVE_ASYNC_LDS
  issue_noise(1, 0);
#endif

  for (int t = 1; t < T_; ++t) {
    const int pb = (t - 1) & 1;
#if HAVE_ASYNC_LDS
    if (t + 1 < T_) issue_noise(t + 1, pb ^ 1);
#endif

    // stage z_prev (acc layout -> row-major [16][32] f16)
#pragma unroll
    for (int n = 0; n < 2; ++n)
#pragma unroll
      for (int v = 0; v < 8; ++v)
        zst[wv][v + 8 * g][16 * n + cl] = (_Float16)zacc[n][v];
    asm volatile("s_wait_dscnt 0" ::: "memory");

    v16h a = frag_from_lds_row(&zst[wv][cl][0], g);

    // h_pre = z_prev @ W1 + b1   (8 tiles of 16x16)
    v8f hacc[8];
#pragma unroll
    for (int n = 0; n < 8; ++n) {
      v8f hseed;
#pragma unroll
      for (int i = 0; i < 8; ++i) hseed[i] = b1v[n];
      hacc[n] = wmma32(a, w1f[n], hseed);
    }

    // tanh + stage h -> [16][128] f16
#pragma unroll
    for (int n = 0; n < 8; ++n)
#pragma unroll
      for (int v = 0; v < 8; ++v)
        hst[wv][v + 8 * g][16 * n + cl] = (_Float16)tanhf(hacc[n][v]);
    asm volatile("s_wait_dscnt 0" ::: "memory");

    v16h a2[4];
#pragma unroll
    for (int kc = 0; kc < 4; ++kc)
      a2[kc] = frag_from_lds_row(&hst[wv][cl][32 * kc], g);

#if HAVE_ASYNC_LDS
    // Retire this step's 12 async copies; leave next step's 12 in flight.
    if (t + 1 < T_) asm volatile("s_wait_asynccnt 12" ::: "memory");
    else            asm volatile("s_wait_asynccnt 0" ::: "memory");
#endif

    // z_t = tanh(..) @ W2 + b2 + Qs * w_t  (seed accumulator with bias+noise)
#pragma unroll
    for (int n = 0; n < 2; ++n) {
      int col = 16 * n + cl;
      v8f c;
#pragma unroll
      for (int v = 0; v < 8; ++v) {
        int row = v + 8 * g;
#if HAVE_ASYNC_LDS
        float w = nbuf[wv][pb][0][row][col] +
                  nbuf[wv][pb][1][row][col] * nbuf[wv][pb][2][row][col];
#else
        int b = b0 + row;
        size_t bt = (size_t)b * T_ + t;
        size_t sbt = ((size_t)(s * B_ + b)) * T_ + t;
        float w =
            m_ws[bt * L_ + col] + sp_ws[bt * L_ + col] * eps[sbt * L_ + col];
#endif
        c[v] = b2v[n] + Qs[n] * w;
      }
#pragma unroll
      for (int kc = 0; kc < 4; ++kc) c = wmma32(a2[kc], w2f[kc][n], c);
      zacc[n] = c;
#pragma unroll
      for (int v = 0; v < 8; ++v) {
        int b = b0 + v + 8 * g;
        zout[(((size_t)(s * B_ + b)) * T_ + t) * L_ + col] = c[v];
      }
    }
  }
}

// ---------------- Kernel 3: Poisson readout z@C + expected log-lik ----------------
// 32000 row tiles of 16 over flattened (s,b,t); one wave per tile, 8 WMMAs (K=32).
__global__ __launch_bounds__(128) void k_readout(
    const float* __restrict__ zsrc, const float* __restrict__ y,
    const float* __restrict__ Cm, const float* __restrict__ dvec,
    float* __restrict__ loss) {
  __shared__ _Float16 zst[4][16][32];
  const int lane = threadIdx.x & 31;
  const int wv = threadIdx.x >> 5;
  const int g = lane >> 4;
  const int cl = lane & 15;
  const int tile = blockIdx.x * 4 + wv;  // 0..31999
  const int R0 = tile * 16;

  v16h cf[8];
#pragma unroll
  for (int n = 0; n < 8; ++n) cf[n] = bfrag_global(Cm, N_, 0, 16 * n + cl, g);
  float db[8];
#pragma unroll
  for (int n = 0; n < 8; ++n) db[n] = dvec[16 * n + cl];

  // stage z tile (contiguous [16][32] f32) -> f16 LDS
#pragma unroll
  for (int i = 0; i < 16; ++i) {
    int idx = lane + 32 * i;
    zst[wv][idx >> 5][idx & 31] = (_Float16)zsrc[(size_t)R0 * L_ + idx];
  }
  asm volatile("s_wait_dscnt 0" ::: "memory");
  v16h a = frag_from_lds_row(&zst[wv][cl][0], g);

  size_t ybase[8];
#pragma unroll
  for (int v = 0; v < 8; ++v) {
    int gr = R0 + v + 8 * g;  // global (s,b,t) row
    int bt = gr % BT_;        // y row index
    ybase[v] = (size_t)bt * N_;
  }

  float p = 0.f;
#pragma unroll
  for (int n = 0; n < 8; ++n) {
    v8f c;
#pragma unroll
    for (int i = 0; i < 8; ++i) c[i] = db[n];
    c = wmma32(a, cf[n], c);
    int col = 16 * n + cl;
#pragma unroll
    for (int v = 0; v < 8; ++v) {
      float lr = c[v];
      p += y[ybase[v] + col] * lr - expf(lr);
    }
  }
#pragma unroll
  for (int o = 16; o > 0; o >>= 1) p += __shfl_xor(p, o, 32);
  if (lane == 0) atomicAdd(loss, -p * (1.0f / (B_ * S_)));
}

// ---------------- Kernel 4: z_mean over samples ----------------
__global__ __launch_bounds__(256) void k_zmean(const float* __restrict__ z,
                                               float* __restrict__ zm) {
  int i = blockIdx.x * 256 + threadIdx.x;
  if (i < BTL_) {
    float sum = 0.f;
#pragma unroll
    for (int s = 0; s < S_; ++s) sum += z[(size_t)s * BTL_ + i];
    zm[i] = sum * 0.125f;
  }
}

extern "C" void kernel_launch(void* const* d_in, const int* in_sizes, int n_in,
                              void* d_out, int out_size, void* d_ws,
                              size_t ws_size, hipStream_t stream) {
  const float* y      = (const float*)d_in[0];
  const float* eps    = (const float*)d_in[1];
  const float* tmask  = (const float*)d_in[2];
  const float* enc_Wm = (const float*)d_in[3];
  const float* enc_bm = (const float*)d_in[4];
  const float* enc_WP = (const float*)d_in[5];
  const float* enc_bP = (const float*)d_in[6];
  const float* dyn_W1 = (const float*)d_in[7];
  const float* dyn_b1 = (const float*)d_in[8];
  const float* dyn_W2 = (const float*)d_in[9];
  const float* dyn_b2 = (const float*)d_in[10];
  const float* log_Q  = (const float*)d_in[11];
  const float* m_0    = (const float*)d_in[12];
  const float* log_Q0 = (const float*)d_in[13];
  const float* Cmat   = (const float*)d_in[14];
  const float* dvec   = (const float*)d_in[15];

  float* out   = (float*)d_out;
  float* loss  = out;            // [1]
  float* zout  = out + 1;        // [S,B,T,L]
  float* zmean = zout + SBTL_;   // [B,T,L]

  float* m_ws  = (float*)d_ws;   // [B,T,L]
  float* sp_ws = m_ws + BTL_;    // [B,T,L]

  k_init_loss<<<1, 32, 0, stream>>>(loss);
  // encoder: 4000 tiles, 4 waves/block
  k_encoder<<<1000, 128, 0, stream>>>(y, tmask, enc_Wm, enc_bm, enc_WP, enc_bP,
                                      m_ws, sp_ws, loss);
  // scan: 32 tiles, 2 waves/block (LDS double-buffer for async noise staging)
  k_scan<<<16, 64, 0, stream>>>(eps, dyn_W1, dyn_b1, dyn_W2, dyn_b2, log_Q,
                                m_0, log_Q0, m_ws, sp_ws, zout);
  // readout: 32000 tiles, 4 waves/block
  k_readout<<<8000, 128, 0, stream>>>(zout, y, Cmat, dvec, loss);
  // mean over samples
  k_zmean<<<(BTL_ + 255) / 256, 256, 0, stream>>>(zout, zmean);
}
